// Seq2Seq_33285996544699
// MI455X (gfx1250) — matrix operand order repaired and checked by reading, not compiled, not run on hardware
//
#include <hip/hip_runtime.h>
#include <hip/hip_bf16.h>

typedef __bf16 bf16;
typedef __attribute__((ext_vector_type(16))) bf16  v16bf;
typedef __attribute__((ext_vector_type(8)))  bf16  v8bf;
typedef __attribute__((ext_vector_type(8)))  float v8f;

#define BSZ   64
#define SEQ_S 64
#define SEQ_T 32
#define TDEC  31
#define EDIM  256
#define HDIM  512
#define G4    2048
#define VOC   32000

#define LDA_P 40    // 64x32 bf16 A tile, padded row stride (80B rows -> 16B aligned chunks)

__device__ __forceinline__ float sigf(float x) { return 1.0f / (1.0f + __expf(-x)); }

// A fragment: 16x32 bf16 (rows row0..row0+15 of LDS A tile).
// Lane l: row M=l&15, K=half*16..+7 (VGPR0-3) and K=half*16+8..+15 (VGPR4-7).
// Both chunks are 16B-contiguous in LDS -> two ds_load_b128.
__device__ __forceinline__ v16bf frag_a(const bf16* As, int row0, int lane) {
  const int half = lane >> 4, m = lane & 15;
  const bf16* p = As + (row0 + m) * LDA_P;
  v8bf lo = *(const v8bf*)(p + half * 16);
  v8bf hi = *(const v8bf*)(p + half * 16 + 8);
  return __builtin_shufflevector(lo, hi, 0, 1, 2, 3, 4, 5, 6, 7,
                                 8, 9, 10, 11, 12, 13, 14, 15);
}

// B fragment from pre-swizzled weights: fragment (nt,kt) stores lane l's
// 16 bf16 (col n = nt*16 + (l&15), K = kt*32 + (l>>4)*16 .. +15) contiguously.
// -> two global_load_b128 per fragment, zero LDS.
__device__ __forceinline__ v16bf load_bfrag(const bf16* __restrict__ W, int KT,
                                            int nt, int kt, int lane) {
  const v8bf* p = (const v8bf*)(W + ((size_t)((size_t)nt * KT + kt) * 32 + lane) * 16);
  v8bf lo = p[0];
  v8bf hi = p[1];
  return __builtin_shufflevector(lo, hi, 0, 1, 2, 3, 4, 5, 6, 7,
                                 8, 9, 10, 11, 12, 13, 14, 15);
}

// f32 weights [N][K] -> bf16 swizzled fragment-ready layout.
// permute=1 applies the gate-interleave used by the LSTM step blocks:
// permuted col n -> orig col ((nl>>5)*512 + (n>>7)*32 + (nl&31), nl=n&127).
__global__ void swizzle_weight_kernel(const float* __restrict__ in,
                                      bf16* __restrict__ out,
                                      int N, int K, int permute) {
  size_t idx = (size_t)blockIdx.x * 256 + threadIdx.x;
  size_t total = (size_t)N * K;
  if (idx >= total) return;
  const int KT = K >> 5;
  int kr   = (int)(idx & 15);
  int lane = (int)((idx >> 4) & 31);
  int ft   = (int)(idx >> 9);
  int kt = ft % KT, nt = ft / KT;
  int n = nt * 16 + (lane & 15);
  int k = kt * 32 + (lane >> 4) * 16 + kr;
  int col = n;
  if (permute) {
    int blk = n >> 7, nl = n & 127;
    col = (nl >> 5) * HDIM + blk * 32 + (nl & 31);
  }
  out[idx] = (bf16)in[(size_t)col * K + k];
}

__global__ void zero_f32_kernel(float* __restrict__ p, size_t n) {
  size_t idx = (size_t)blockIdx.x * 256 + threadIdx.x;
  if (idx < n) p[idx] = 0.0f;
}

__global__ void zero_row0_kernel(float* __restrict__ out) {
  size_t idx = (size_t)blockIdx.x * 256 + threadIdx.x;
  if (idx >= (size_t)BSZ * VOC) return;
  int b = (int)(idx / VOC);
  int v = (int)(idx % VOC);
  out[(size_t)b * SEQ_T * VOC + v] = 0.0f;
}

// xp[t*B+b][g] = emb[tok(b,t)] . W_ih[g,:] + b_ih[g] + b_hh[g]
// Block tile 64x128, K step 32, 8 waves (2x4), each wave 32x32 (4 WMMA accs).
__global__ __launch_bounds__(256) void xp_gemm_kernel(
    const int* __restrict__ toks, int tokStride,
    const float* __restrict__ emb,
    const bf16* __restrict__ Wswz,    // swizzled [NT][KT][32][16]
    const float* __restrict__ bih, const float* __restrict__ bhh,
    float* __restrict__ xp) {
  __shared__ bf16 As[64 * LDA_P];
  __shared__ int  tok[64];

  const int tid = threadIdx.x;
  const int lane = tid & 31, w = tid >> 5;
  const int wm = w >> 2, wn = w & 3;
  const int m0 = blockIdx.y * 64, n0 = blockIdx.x * 128;
  const int KT = EDIM >> 5;
  const int ntB = (n0 >> 4) + 2 * wn;

  if (tid < 64) {
    int r = m0 + tid;
    int t = r >> 6, b = r & 63;   // r = t*64 + b
    tok[tid] = toks[b * tokStride + t];
  }
  __syncthreads();

  v8f acc00{}, acc01{}, acc10{}, acc11{};
  const int arow = tid >> 2, acs = (tid & 3) << 3;
  const float* abase = emb + (size_t)tok[arow] * EDIM + acs;  // loop-invariant gather base
  bf16* asl = As + arow * LDA_P + acs;

  for (int k0 = 0; k0 < EDIM; k0 += 32) {
    const float* ap = abase + k0;
    v8bf av;
#pragma unroll
    for (int e = 0; e < 8; ++e) av[e] = (bf16)ap[e];
    *(v8bf*)asl = av;
    __syncthreads();

    const int kt = k0 >> 5;
    v16bf a0 = frag_a(As, 32 * wm, lane);
    v16bf a1 = frag_a(As, 32 * wm + 16, lane);
    v16bf b0 = load_bfrag(Wswz, KT, ntB, kt, lane);
    v16bf b1 = load_bfrag(Wswz, KT, ntB + 1, kt, lane);
    acc00 = __builtin_amdgcn_wmma_f32_16x16x32_bf16(false, a0, false, b0, (short)0, acc00, false, false);
    acc01 = __builtin_amdgcn_wmma_f32_16x16x32_bf16(false, a0, false, b1, (short)0, acc01, false, false);
    acc10 = __builtin_amdgcn_wmma_f32_16x16x32_bf16(false, a1, false, b0, (short)0, acc10, false, false);
    acc11 = __builtin_amdgcn_wmma_f32_16x16x32_bf16(false, a1, false, b1, (short)0, acc11, false, false);
    __syncthreads();
  }

  const int half = lane >> 4, nl = lane & 15;
  v8f accs[2][2] = {{acc00, acc01}, {acc10, acc11}};
#pragma unroll
  for (int i = 0; i < 2; ++i)
#pragma unroll
    for (int jj = 0; jj < 2; ++jj)
#pragma unroll
      for (int r = 0; r < 8; ++r) {
        int row = m0 + 32 * wm + 16 * i + half * 8 + r;
        int col = n0 + 32 * wn + 16 * jj + nl;
        xp[(size_t)row * G4 + col] = accs[i][jj][r] + bih[col] + bhh[col];
      }
}

// One LSTM step, fused: gates = xp_t + h_in @ Whh^T (gate-permuted swizzle),
// then sigmoid/tanh state update. Block = 64 batches x (4 gates x 32 hidden).
__global__ __launch_bounds__(256) void lstm_step_kernel(
    const float* __restrict__ xp_t,   // [64][2048]
    const bf16* __restrict__ Wswz,    // swizzled+permuted [128 nt][16 kt][32][16]
    const float* __restrict__ h_in,   // [64][512]
    float* __restrict__ h_out,        // [64][512]
    float* __restrict__ cst,          // [64][512] in/out
    bf16* __restrict__ hseq_bf,       // [B*31][512] or null
    int t) {
  __shared__ bf16  As[64 * LDA_P];
  __shared__ float gsh[4 * 64 * 33];

  const int tid = threadIdx.x;
  const int lane = tid & 31, w = tid >> 5;
  const int wm = w >> 2, wn = w & 3;
  const int j0 = blockIdx.x * 32;
  const int KT = HDIM >> 5;
  const int ntB = blockIdx.x * 8 + 2 * wn;   // permuted N-order fragments

  v8f acc00{}, acc01{}, acc10{}, acc11{};
  const int arow = tid >> 2, acs = (tid & 3) << 3;
  const float* abase = h_in + (size_t)arow * HDIM + acs;
  bf16* asl = As + arow * LDA_P + acs;

  for (int k0 = 0; k0 < HDIM; k0 += 32) {
    const float* ap = abase + k0;
    v8bf av;
#pragma unroll
    for (int e = 0; e < 8; ++e) av[e] = (bf16)ap[e];
    *(v8bf*)asl = av;
    __syncthreads();

    const int kt = k0 >> 5;
    v16bf a0 = frag_a(As, 32 * wm, lane);
    v16bf a1 = frag_a(As, 32 * wm + 16, lane);
    v16bf b0 = load_bfrag(Wswz, KT, ntB, kt, lane);
    v16bf b1 = load_bfrag(Wswz, KT, ntB + 1, kt, lane);
    acc00 = __builtin_amdgcn_wmma_f32_16x16x32_bf16(false, a0, false, b0, (short)0, acc00, false, false);
    acc01 = __builtin_amdgcn_wmma_f32_16x16x32_bf16(false, a0, false, b1, (short)0, acc01, false, false);
    acc10 = __builtin_amdgcn_wmma_f32_16x16x32_bf16(false, a1, false, b0, (short)0, acc10, false, false);
    acc11 = __builtin_amdgcn_wmma_f32_16x16x32_bf16(false, a1, false, b1, (short)0, acc11, false, false);
    __syncthreads();
  }

  const int half = lane >> 4, nl = lane & 15;
  v8f accs[2][2] = {{acc00, acc01}, {acc10, acc11}};
#pragma unroll
  for (int i = 0; i < 2; ++i)
#pragma unroll
    for (int jj = 0; jj < 2; ++jj)
#pragma unroll
      for (int r = 0; r < 8; ++r) {
        int m = 32 * wm + 16 * i + half * 8 + r;       // batch
        int jloc = 16 * jj + nl;                       // hidden unit in slice
        int col = wn * HDIM + j0 + jloc;               // gate wn, hidden j0+jloc
        gsh[(wn * 64 + m) * 33 + jloc] = accs[i][jj][r] + xp_t[(size_t)m * G4 + col];
      }
  __syncthreads();

#pragma unroll
  for (int e = 0; e < 8; ++e) {
    int p = tid + (e << 8);    // 0..2047
    int b = p >> 5, j = p & 31;
    float ig = gsh[(0 * 64 + b) * 33 + j];
    float fg = gsh[(1 * 64 + b) * 33 + j];
    float gg = gsh[(2 * 64 + b) * 33 + j];
    float og = gsh[(3 * 64 + b) * 33 + j];
    float cv = cst[(size_t)b * HDIM + j0 + j];
    float cn = sigf(fg) * cv + sigf(ig) * tanhf(gg);
    float hn = sigf(og) * tanhf(cn);
    cst[(size_t)b * HDIM + j0 + j] = cn;
    h_out[(size_t)b * HDIM + j0 + j] = hn;
    if (hseq_bf) hseq_bf[((size_t)b * TDEC + t) * HDIM + j0 + j] = (bf16)hn;
  }
}

// logits: out[b][t+1][v] = h_seq[b][t] . fc_W[v,:] + fc_b[v]
__global__ __launch_bounds__(256) void logits_gemm_kernel(
    const bf16* __restrict__ hseq,    // [1984][512]
    const bf16* __restrict__ Wswz,    // swizzled [2000 nt][16 kt][32][16]
    const float* __restrict__ fcb,
    float* __restrict__ out) {
  __shared__ bf16 As[64 * LDA_P];

  const int tid = threadIdx.x;
  const int lane = tid & 31, w = tid >> 5;
  const int wm = w >> 2, wn = w & 3;
  const int m0 = blockIdx.y * 64, n0 = blockIdx.x * 128;
  const int KT = HDIM >> 5;
  const int ntB = (n0 >> 4) + 2 * wn;

  v8f acc00{}, acc01{}, acc10{}, acc11{};
  const int arow = tid >> 2, acs = (tid & 3) << 3;
  const bf16* abase = hseq + (size_t)(m0 + arow) * HDIM + acs;
  bf16* asl = As + arow * LDA_P + acs;

  for (int k0 = 0; k0 < HDIM; k0 += 32) {
    *(v8bf*)asl = *(const v8bf*)(abase + k0);
    __syncthreads();

    const int kt = k0 >> 5;
    v16bf a0 = frag_a(As, 32 * wm, lane);
    v16bf a1 = frag_a(As, 32 * wm + 16, lane);
    v16bf b0 = load_bfrag(Wswz, KT, ntB, kt, lane);
    v16bf b1 = load_bfrag(Wswz, KT, ntB + 1, kt, lane);
    acc00 = __builtin_amdgcn_wmma_f32_16x16x32_bf16(false, a0, false, b0, (short)0, acc00, false, false);
    acc01 = __builtin_amdgcn_wmma_f32_16x16x32_bf16(false, a0, false, b1, (short)0, acc01, false, false);
    acc10 = __builtin_amdgcn_wmma_f32_16x16x32_bf16(false, a1, false, b0, (short)0, acc10, false, false);
    acc11 = __builtin_amdgcn_wmma_f32_16x16x32_bf16(false, a1, false, b1, (short)0, acc11, false, false);
    __syncthreads();
  }

  const int half = lane >> 4, nl = lane & 15;
  v8f accs[2][2] = {{acc00, acc01}, {acc10, acc11}};
#pragma unroll
  for (int i = 0; i < 2; ++i)
#pragma unroll
    for (int jj = 0; jj < 2; ++jj)
#pragma unroll
      for (int r = 0; r < 8; ++r) {
        int row = m0 + 32 * wm + 16 * i + half * 8 + r;  // row = b*31 + t
        int b = row / TDEC, t = row % TDEC;
        int col = n0 + 32 * wn + 16 * jj + nl;
        out[((size_t)b * SEQ_T + t + 1) * VOC + col] = accs[i][jj][r] + fcb[col];
      }
}

extern "C" void kernel_launch(void* const* d_in, const int* in_sizes, int n_in,
                              void* d_out, int out_size, void* d_ws, size_t ws_size,
                              hipStream_t stream) {
  (void)in_sizes; (void)n_in; (void)out_size; (void)ws_size;
  const int*   src     = (const int*)d_in[0];
  const int*   tgt     = (const int*)d_in[1];
  const float* enc_emb = (const float*)d_in[2];
  const float* dec_emb = (const float*)d_in[3];
  const float* enc_Wih = (const float*)d_in[4];
  const float* enc_Whh = (const float*)d_in[5];
  const float* enc_bih = (const float*)d_in[6];
  const float* enc_bhh = (const float*)d_in[7];
  const float* dec_Wih = (const float*)d_in[8];
  const float* dec_Whh = (const float*)d_in[9];
  const float* dec_bih = (const float*)d_in[10];
  const float* dec_bhh = (const float*)d_in[11];
  const float* fc_W    = (const float*)d_in[12];
  const float* fc_b    = (const float*)d_in[13];
  float* out = (float*)d_out;

  char* ws = (char*)d_ws;
  size_t off = 0;
  auto alloc = [&](size_t bytes) -> char* {
    char* p = ws + off;
    off += (bytes + 255) & ~(size_t)255;
    return p;
  };
  bf16*  encWihS = (bf16*)alloc((size_t)EDIM * G4 * 2);
  bf16*  decWihS = (bf16*)alloc((size_t)EDIM * G4 * 2);
  bf16*  encWhhS = (bf16*)alloc((size_t)HDIM * G4 * 2);
  bf16*  decWhhS = (bf16*)alloc((size_t)HDIM * G4 * 2);
  bf16*  fcWS    = (bf16*)alloc((size_t)HDIM * VOC * 2);
  float* xp_enc  = (float*)alloc((size_t)SEQ_S * BSZ * G4 * 4);
  float* xp_dec  = (float*)alloc((size_t)TDEC * BSZ * G4 * 4);
  float* hA      = (float*)alloc((size_t)BSZ * HDIM * 4);
  float* hB      = (float*)alloc((size_t)BSZ * HDIM * 4);
  float* cS      = (float*)alloc((size_t)BSZ * HDIM * 4);
  bf16*  hseq    = (bf16*)alloc((size_t)BSZ * TDEC * HDIM * 2);

  auto cdiv = [](size_t a, size_t b) -> unsigned { return (unsigned)((a + b - 1) / b); };

  // bf16 weight swizzles into WMMA-fragment-ready layout
  swizzle_weight_kernel<<<cdiv((size_t)G4 * EDIM, 256), 256, 0, stream>>>(enc_Wih, encWihS, G4, EDIM, 0);
  swizzle_weight_kernel<<<cdiv((size_t)G4 * EDIM, 256), 256, 0, stream>>>(dec_Wih, decWihS, G4, EDIM, 0);
  swizzle_weight_kernel<<<cdiv((size_t)G4 * HDIM, 256), 256, 0, stream>>>(enc_Whh, encWhhS, G4, HDIM, 1);
  swizzle_weight_kernel<<<cdiv((size_t)G4 * HDIM, 256), 256, 0, stream>>>(dec_Whh, decWhhS, G4, HDIM, 1);
  swizzle_weight_kernel<<<cdiv((size_t)VOC * HDIM, 256), 256, 0, stream>>>(fc_W, fcWS, VOC, HDIM, 0);

  // state init + output row-0 zeros (d_out is poisoned)
  zero_f32_kernel<<<cdiv((size_t)BSZ * HDIM, 256), 256, 0, stream>>>(hA, (size_t)BSZ * HDIM);
  zero_f32_kernel<<<cdiv((size_t)BSZ * HDIM, 256), 256, 0, stream>>>(cS, (size_t)BSZ * HDIM);
  zero_row0_kernel<<<cdiv((size_t)BSZ * VOC, 256), 256, 0, stream>>>(out);

  // input-projection GEMMs (embedding gathered in-kernel)
  xp_gemm_kernel<<<dim3(G4 / 128, (SEQ_S * BSZ) / 64), 256, 0, stream>>>(
      src, SEQ_S, enc_emb, encWihS, enc_bih, enc_bhh, xp_enc);
  xp_gemm_kernel<<<dim3(G4 / 128, (TDEC * BSZ) / 64), 256, 0, stream>>>(
      tgt, SEQ_T, dec_emb, decWihS, dec_bih, dec_bhh, xp_dec);

  // sequential recurrence, h double-buffered (cross-block RAW on h)
  float* hc = hA; float* hn = hB;
  for (int t = 0; t < SEQ_S; ++t) {
    lstm_step_kernel<<<HDIM / 32, 256, 0, stream>>>(
        xp_enc + (size_t)t * BSZ * G4, encWhhS, hc, hn, cS, (bf16*)nullptr, t);
    float* tmp = hc; hc = hn; hn = tmp;
  }
  for (int t = 0; t < TDEC; ++t) {
    lstm_step_kernel<<<HDIM / 32, 256, 0, stream>>>(
        xp_dec + (size_t)t * BSZ * G4, decWhhS, hc, hn, cS, hseq, t);
    float* tmp = hc; hc = hn; hn = tmp;
  }

  // final vocabulary projection (output-bandwidth bound)
  logits_gemm_kernel<<<dim3(VOC / 128, (BSZ * TDEC) / 64), 256, 0, stream>>>(
      hseq, fcWS, fc_b, out);
}
